// ShiftedWindowMSA_46196668236434
// MI455X (gfx1250) — matrix operand order repaired and checked by reading, not compiled, not run on hardware
//
#include <hip/hip_runtime.h>
#include <hip/hip_bf16.h>

typedef __attribute__((ext_vector_type(16))) _Float16 v16h;
typedef __attribute__((ext_vector_type(8)))  _Float16 v8h;
typedef __attribute__((ext_vector_type(8)))  float    v8f;
typedef __attribute__((ext_vector_type(4)))  int      v4i;
typedef unsigned long long ull;

#if defined(__has_builtin)
#if __has_builtin(__builtin_amdgcn_global_load_async_to_lds_b128) && \
    __has_builtin(__builtin_amdgcn_s_wait_asynccnt)
#define HAVE_ASYNC_LDS 1
#endif
#endif
#ifndef HAVE_ASYNC_LDS
#define HAVE_ASYNC_LDS 0
#endif

#if HAVE_ASYNC_LDS
typedef __attribute__((address_space(1))) v4i* gptr_b128;
typedef __attribute__((address_space(3))) v4i* lptr_b128;
#endif

#define WMMA_F16(a, b, c) \
  __builtin_amdgcn_wmma_f32_16x16x32_f16(false, (a), false, (b), (short)0, (c), false, false)

static constexpr int Bn = 2, Tn = 8, Hn = 128, Wn = 192, Cn = 96;
static constexpr int HWn = Hn * Wn;   // 24576
static constexpr int NWIN = 16 * 24;  // 384 windows per (b,t)

__device__ __forceinline__ v8h zero8h() {
  v8h z = {(_Float16)0, (_Float16)0, (_Float16)0, (_Float16)0,
           (_Float16)0, (_Float16)0, (_Float16)0, (_Float16)0};
  return z;
}
__device__ __forceinline__ v8f zero8f() {
  v8f z = {0.f, 0.f, 0.f, 0.f, 0.f, 0.f, 0.f, 0.f};
  return z;
}

// A-fragment (16x32 f16) from row-major f16 tile, full K=32 valid.
__device__ __forceinline__ v16h load_a16(const _Float16* src, int stride,
                                         int r0, int k0) {
  const int l = threadIdx.x & 31;
  const _Float16* p = src + (r0 + (l & 15)) * stride + k0 + ((l >> 4) * 8);
  v8h lo = *(const v8h*)p;
  v8h hi = *(const v8h*)(p + 16);
  return __builtin_shufflevector(lo, hi, 0, 1, 2, 3, 4, 5, 6, 7,
                                 8, 9, 10, 11, 12, 13, 14, 15);
}

// A-fragment with K=16 valid (upper half zero).
__device__ __forceinline__ v16h load_a16_k16(const _Float16* src, int stride,
                                             int r0) {
  const int l = threadIdx.x & 31;
  const _Float16* p = src + (r0 + (l & 15)) * stride + ((l >> 4) * 8);
  v8h lo = *(const v8h*)p;
  return __builtin_shufflevector(lo, zero8h(), 0, 1, 2, 3, 4, 5, 6, 7,
                                 8, 9, 10, 11, 12, 13, 14, 15);
}

// B-fragment from pre-packed register image: [frag][lane][16 halves].
__device__ __forceinline__ v16h load_b_pk(const _Float16* pk, int frag) {
  return ((const v16h*)pk)[frag * 32 + (threadIdx.x & 31)];
}

// Pack f32 weight W[K][N] into B-fragment register image (chunk = one lane's
// 16 contiguous halves; index math amortized; stores merge to b128).
__device__ void pack_weight(_Float16* dst, const float* __restrict__ W,
                            int K, int N, int idx, int nthreads) {
  const int NT = N / 16;
  const int KT = (K + 31) / 32;
  const int chunks = KT * NT * 32;
  for (int ch = idx; ch < chunks; ch += nthreads) {
    const int frag = ch >> 5, lane = ch & 31;
    const int kt = frag / NT, nt = frag % NT;
    const int kb = kt * 32 + ((lane >> 4) << 3);
    const int n  = nt * 16 + (lane & 15);
    _Float16* d = dst + ch * 16;
    const float* src = W + n;
#pragma unroll
    for (int i = 0; i < 16; ++i) {
      const int k = kb + i + (i & 8);
      d[i] = (_Float16)((k < K) ? src[k * N] : 0.0f);
    }
  }
}

// ===========================================================================
// Kernel 1: LN1 + shifted 8x8 window MHA   (workspace output is f16)
//   0 rb[512] | 512 Xs f16[64][96] | 12800 Wpk 15 frags | 28160 Qs | 30208 Ks
//   32256 Vt f16[16][64] | 34304 Ss f32[2][64][64] | 67072 Ap f16[2][64][64]
//   83456 Os f16[64][16] | 85504 Rcp f32[2][64] | total 86016
// ===========================================================================
__global__ __launch_bounds__(128) void k_spatial_msa(
    const float* __restrict__ x, const float* __restrict__ g1,
    const float* __restrict__ b1, const float* __restrict__ Wq,
    const float* __restrict__ bq, const float* __restrict__ Wk,
    const float* __restrict__ bk, const float* __restrict__ Wv,
    const float* __restrict__ bv, const float* __restrict__ Wo,
    const float* __restrict__ bo, _Float16* __restrict__ xout) {
  extern __shared__ char smem[];
  ull*      rb  = (ull*)smem;
  _Float16* Xs  = (_Float16*)(smem + 512);
  _Float16* Wpk = (_Float16*)(smem + 12800);
  _Float16* Qpk = Wpk;
  _Float16* Kpk = Wpk + 3 * 512;
  _Float16* Vpk = Wpk + 6 * 512;
  _Float16* Opk = Wpk + 9 * 512;
  _Float16* Qs  = (_Float16*)(smem + 28160);
  _Float16* Ks  = (_Float16*)(smem + 30208);
  _Float16* Vt  = (_Float16*)(smem + 32256);
  float*    Ss  = (float*)(smem + 34304);
  _Float16* Ap  = (_Float16*)(smem + 67072);
  _Float16* Os  = (_Float16*)(smem + 83456);
  float*    Rcp = (float*)(smem + 85504);

  const int tid  = threadIdx.x;
  const int wave = tid >> 5;
  const int r0   = wave * 16;
  const int l    = tid & 31, nn = l & 15, mo = (l >> 4) * 8;

  const int wid = blockIdx.x;
  const int bt  = wid / NWIN, win = wid % NWIN;
  const int b   = bt / Tn,    t   = bt % Tn;
  const int wh  = win / 24,   ww  = win % 24;

  __builtin_prefetch(Wq, 0, 0);
  __builtin_prefetch(Wo, 0, 0);

  // ---- Phase A: shifted gather + LN (two-pass, no spill arrays)
  if (tid < 64) {
    const int mh = tid >> 3, mw = tid & 7;
    const int h = (wh * 8 + mh + 4) & (Hn - 1);
    const int w = (ww * 8 + mw + 4) % Wn;
    const size_t base = ((((size_t)b * Tn + t) * Hn + h) * Wn + w) * Cn;
    rb[tid] = base;
    const float4* xr = (const float4*)(x + base);
    float s = 0.f, ss = 0.f;
    for (int c4 = 0; c4 < 24; ++c4) {
      float4 v = xr[c4];
      s  += v.x + v.y + v.z + v.w;
      ss += v.x * v.x + v.y * v.y + v.z * v.z + v.w * v.w;
    }
    const float mu  = s * (1.0f / Cn);
    const float var = ss * (1.0f / Cn) - mu * mu;
    const float inv = rsqrtf(var + 1e-3f);
    for (int c4 = 0; c4 < 24; ++c4) {
      float4 v = xr[c4];  // cache hit
      const int c = c4 * 4;
      Xs[tid * Cn + c + 0] = (_Float16)((v.x - mu) * inv * g1[c + 0] + b1[c + 0]);
      Xs[tid * Cn + c + 1] = (_Float16)((v.y - mu) * inv * g1[c + 1] + b1[c + 1]);
      Xs[tid * Cn + c + 2] = (_Float16)((v.z - mu) * inv * g1[c + 2] + b1[c + 2]);
      Xs[tid * Cn + c + 3] = (_Float16)((v.w - mu) * inv * g1[c + 3] + b1[c + 3]);
    }
  } else {
    const int idx = tid - 64;
    pack_weight(Qpk, Wq, 96, 16, idx, 64);
    pack_weight(Kpk, Wk, 96, 16, idx, 64);
    pack_weight(Vpk, Wv, 96, 16, idx, 64);
    pack_weight(Opk, Wo, 16, 96, idx, 64);
  }
  __syncthreads();

  // ---- Phase B: Q/K/V = LN(x) @ W + b
  {
    v8f aq = zero8f(), ak = zero8f(), av = zero8f();
#pragma unroll
    for (int kt = 0; kt < 3; ++kt) {
      v16h a = load_a16(Xs, 96, r0, kt * 32);
      aq = WMMA_F16(a, load_b_pk(Qpk, kt), aq);
      ak = WMMA_F16(a, load_b_pk(Kpk, kt), ak);
      av = WMMA_F16(a, load_b_pk(Vpk, kt), av);
    }
    const float biq = bq[nn], bik = bk[nn], biv = bv[nn];
#pragma unroll
    for (int i = 0; i < 8; ++i) {
      Qs[(r0 + mo + i) * 16 + nn] = (_Float16)(aq[i] + biq);
      Ks[(r0 + mo + i) * 16 + nn] = (_Float16)(ak[i] + bik);
      Vt[nn * 64 + r0 + mo + i]   = (_Float16)(av[i] + biv);  // transposed
    }
  }
  __syncthreads();

  // ---- Phase C: S = Q @ K^T * scale (per head; K=8, lanes<16 live)
  {
    const float scale = 0.35355339059327373f;
#pragma unroll
    for (int h2 = 0; h2 < 2; ++h2) {
      float* Sh = Ss + h2 * 4096;
      v16h a;
      {
        v8h q = zero8h();
        if (l < 16) q = *(const v8h*)(Qs + (r0 + nn) * 16 + h2 * 8);
        a = __builtin_shufflevector(q, zero8h(), 0, 1, 2, 3, 4, 5, 6, 7,
                                    8, 9, 10, 11, 12, 13, 14, 15);
      }
#pragma unroll
      for (int j = 0; j < 4; ++j) {
        v8h kv = zero8h();
        if (l < 16) kv = *(const v8h*)(Ks + (j * 16 + nn) * 16 + h2 * 8);
        v16h bb = __builtin_shufflevector(kv, zero8h(), 0, 1, 2, 3, 4, 5, 6, 7,
                                          8, 9, 10, 11, 12, 13, 14, 15);
        v8f acc = zero8f();
        acc = WMMA_F16(a, bb, acc);
#pragma unroll
        for (int i = 0; i < 8; ++i)
          Sh[(r0 + mo + i) * 64 + j * 16 + nn] = acc[i] * scale;
      }
    }
  }
  __syncthreads();

  // ---- Phase D: softmax; write UNNORMALIZED exp (f16), defer 1/sum to E
  {
    const int h2 = tid >> 6, r = tid & 63;
    const float* row = Ss + h2 * 4096 + r * 64;
    _Float16* prow = Ap + h2 * 4096 + r * 64;
    float mx = -1e30f;
    for (int i = 0; i < 64; ++i) mx = fmaxf(mx, row[i]);
    float sum = 0.f;
    for (int i = 0; i < 64; ++i) {
      float e = __expf(row[i] - mx);
      prow[i] = (_Float16)e;
      sum += e;
    }
    Rcp[h2 * 64 + r] = 1.0f / sum;
  }
  __syncthreads();

  // ---- Phase E: O = (exp @ V) * rcp_row
  {
#pragma unroll
    for (int h2 = 0; h2 < 2; ++h2) {
      v8f acc = zero8f();
#pragma unroll
      for (int ks = 0; ks < 64; ks += 32) {
        v16h a = load_a16(Ap + h2 * 4096, 64, r0, ks);
        v16h bb;
        if (nn < 8) {
          const _Float16* vp = Vt + (h2 * 8 + nn) * 64 + ks + ((l >> 4) * 8);
          bb = __builtin_shufflevector(*(const v8h*)vp, *(const v8h*)(vp + 16),
                                       0, 1, 2, 3, 4, 5, 6, 7,
                                       8, 9, 10, 11, 12, 13, 14, 15);
        }
        else
          bb = __builtin_shufflevector(zero8h(), zero8h(), 0, 1, 2, 3, 4, 5, 6, 7,
                                       8, 9, 10, 11, 12, 13, 14, 15);
        acc = WMMA_F16(a, bb, acc);
      }
      if (nn < 8) {
#pragma unroll
        for (int i = 0; i < 8; ++i)
          Os[(r0 + mo + i) * 16 + h2 * 8 + nn] =
              (_Float16)(acc[i] * Rcp[h2 * 64 + r0 + mo + i]);
      }
    }
  }
  __syncthreads();

  // ---- Phase F: out = O @ Wo + bo -> f16 workspace (same offsets as load)
  {
    v16h a = load_a16_k16(Os, 16, r0);
#pragma unroll
    for (int j = 0; j < 6; ++j) {
      v8f acc = zero8f();
      acc = WMMA_F16(a, load_b_pk(Opk, j), acc);
      const float bias = bo[j * 16 + nn];
#pragma unroll
      for (int i = 0; i < 8; ++i)
        xout[rb[r0 + mo + i] + j * 16 + nn] = (_Float16)(acc[i] + bias);
    }
  }
}

// ===========================================================================
// Kernel 2: temporal MHA over T=8, in place on f16 workspace; rows gathered
// with async global->LDS (ASYNCcnt) when available.
//   0 rb[512] | 512 Xs f16[64][96] | 12800 Wpk | 28160 Qs | 30208 Ks
//   32256 Vs | 34304 Os | total 36352
// ===========================================================================
__global__ __launch_bounds__(128) void k_temporal_msa(
    const float* __restrict__ Wq, const float* __restrict__ bq,
    const float* __restrict__ Wk, const float* __restrict__ bk,
    const float* __restrict__ Wv, const float* __restrict__ bv,
    const float* __restrict__ Wo, const float* __restrict__ bo,
    _Float16* __restrict__ xbuf) {
  extern __shared__ char smem[];
  ull*      rb  = (ull*)smem;
  _Float16* Xs  = (_Float16*)(smem + 512);
  _Float16* Wpk = (_Float16*)(smem + 12800);
  _Float16* Qpk = Wpk;
  _Float16* Kpk = Wpk + 3 * 512;
  _Float16* Vpk = Wpk + 6 * 512;
  _Float16* Opk = Wpk + 9 * 512;
  _Float16* Qs  = (_Float16*)(smem + 28160);
  _Float16* Ks  = (_Float16*)(smem + 30208);
  _Float16* Vs  = (_Float16*)(smem + 32256);
  _Float16* Os  = (_Float16*)(smem + 34304);

  const int tid  = threadIdx.x;
  const int wave = tid >> 5;
  const int r0   = wave * 16;
  const int l    = tid & 31, nn = l & 15, mo = (l >> 4) * 8;

  const int g    = blockIdx.x * 8;
  const int b    = g / HWn;
  const int loc0 = g % HWn;

  if (tid < 64) {
    const int s = tid >> 3, tt = tid & 7;
    rb[tid] = (((size_t)b * Tn + tt) * HWn + (loc0 + s)) * Cn;
  } else {
    const int idx = tid - 64;
    pack_weight(Qpk, Wq, 96, 16, idx, 64);
    pack_weight(Kpk, Wk, 96, 16, idx, 64);
    pack_weight(Vpk, Wv, 96, 16, idx, 64);
    pack_weight(Opk, Wo, 16, 96, idx, 64);
  }

  // ---- raw f16 gather of 64 rows x 192B, 16B chunks (async -> LDS)
  for (int ch = tid; ch < 64 * 12; ch += 128) {
    const int row = ch / 12, cc = (ch % 12) * 8;
    const int s = row >> 3, tt = row & 7;
    const size_t base =
        (((size_t)b * Tn + tt) * HWn + (loc0 + s)) * Cn + cc;
#if HAVE_ASYNC_LDS
    __builtin_amdgcn_global_load_async_to_lds_b128(
        (gptr_b128)(xbuf + base), (lptr_b128)(Xs + row * 96 + cc), 0, 0);
#else
    *(v8h*)(Xs + row * 96 + cc) = *(const v8h*)(xbuf + base);
#endif
  }
#if HAVE_ASYNC_LDS
  __builtin_amdgcn_s_wait_asynccnt(0);
#endif
  __syncthreads();

  // ---- Q/K/V projections via WMMA
  {
    v8f aq = zero8f(), ak = zero8f(), av = zero8f();
#pragma unroll
    for (int kt = 0; kt < 3; ++kt) {
      v16h a = load_a16(Xs, 96, r0, kt * 32);
      aq = WMMA_F16(a, load_b_pk(Qpk, kt), aq);
      ak = WMMA_F16(a, load_b_pk(Kpk, kt), ak);
      av = WMMA_F16(a, load_b_pk(Vpk, kt), av);
    }
    const float biq = bq[nn], bik = bk[nn], biv = bv[nn];
#pragma unroll
    for (int i = 0; i < 8; ++i) {
      Qs[(r0 + mo + i) * 16 + nn] = (_Float16)(aq[i] + biq);
      Ks[(r0 + mo + i) * 16 + nn] = (_Float16)(ak[i] + bik);
      Vs[(r0 + mo + i) * 16 + nn] = (_Float16)(av[i] + biv);
    }
  }
  __syncthreads();

  // ---- tiny 8x8 attention per (seq, head, query-t): pure VALU
  {
    const int s  = tid >> 4;
    const int h2 = (tid >> 3) & 1;
    const int tq = tid & 7;
    const int qr = s * 8 + tq;
    float q[8], sc[8];
#pragma unroll
    for (int k = 0; k < 8; ++k) q[k] = (float)Qs[qr * 16 + h2 * 8 + k];
    float mx = -1e30f;
#pragma unroll
    for (int tk = 0; tk < 8; ++tk) {
      float d = 0.f;
#pragma unroll
      for (int k = 0; k < 8; ++k)
        d += q[k] * (float)Ks[(s * 8 + tk) * 16 + h2 * 8 + k];
      sc[tk] = d * 0.35355339059327373f;
      mx = fmaxf(mx, sc[tk]);
    }
    float sum = 0.f;
#pragma unroll
    for (int tk = 0; tk < 8; ++tk) { sc[tk] = __expf(sc[tk] - mx); sum += sc[tk]; }
    const float rcp = 1.0f / sum;
#pragma unroll
    for (int k = 0; k < 8; ++k) {
      float o = 0.f;
#pragma unroll
      for (int tk = 0; tk < 8; ++tk)
        o += sc[tk] * (float)Vs[(s * 8 + tk) * 16 + h2 * 8 + k];
      Os[qr * 16 + h2 * 8 + k] = (_Float16)(o * rcp);
    }
  }
  __syncthreads();

  // ---- output projection, write back in place (f16)
  {
    v16h a = load_a16_k16(Os, 16, r0);
#pragma unroll
    for (int j = 0; j < 6; ++j) {
      v8f acc = zero8f();
      acc = WMMA_F16(a, load_b_pk(Opk, j), acc);
      const float bias = bo[j * 16 + nn];
#pragma unroll
      for (int i = 0; i < 8; ++i)
        xbuf[rb[r0 + mo + i] + j * 16 + nn] = (_Float16)(acc[i] + bias);
    }
  }
}

// ===========================================================================
// Kernel 3: LN2 + MLP (two 96x96 matmuls, ReLU), f16 in -> f32 out
//   0 Xs f16[64][96] | 12288 Wpk 18 frags | 30720 Hb f16[64][96] | 43008 total
// ===========================================================================
__global__ __launch_bounds__(128) void k_mlp(
    const _Float16* __restrict__ xbuf, const float* __restrict__ g2,
    const float* __restrict__ b2, const float* __restrict__ W1,
    const float* __restrict__ b1m, const float* __restrict__ W2,
    const float* __restrict__ b2m, float* __restrict__ out) {
  extern __shared__ char smem[];
  _Float16* Xs  = (_Float16*)smem;
  _Float16* Wpk = (_Float16*)(smem + 12288);
  _Float16* Hb  = (_Float16*)(smem + 30720);

  const int tid  = threadIdx.x;
  const int wave = tid >> 5;
  const int r0   = wave * 16;
  const int l    = tid & 31, nn = l & 15, mo = (l >> 4) * 8;
  const int tok0 = blockIdx.x * 64;

  if (tid < 64) {
    const size_t base = (size_t)(tok0 + tid) * Cn;
    const v8h* xr = (const v8h*)(xbuf + base);
    float s = 0.f, ss = 0.f;
    for (int c8 = 0; c8 < 12; ++c8) {
      v8h v = xr[c8];
#pragma unroll
      for (int j = 0; j < 8; ++j) {
        float f = (float)v[j];
        s += f; ss += f * f;
      }
    }
    const float mu  = s * (1.0f / Cn);
    const float var = ss * (1.0f / Cn) - mu * mu;
    const float inv = rsqrtf(var + 1e-3f);
    for (int c8 = 0; c8 < 12; ++c8) {
      v8h v = xr[c8];  // cache hit
      const int c = c8 * 8;
#pragma unroll
      for (int j = 0; j < 8; ++j)
        Xs[tid * Cn + c + j] =
            (_Float16)(((float)v[j] - mu) * inv * g2[c + j] + b2[c + j]);
    }
  } else {
    pack_weight(Wpk, W1, 96, 96, tid - 64, 64);
  }
  __syncthreads();

  // ---- H = relu(X @ W1 + b1m)
#pragma unroll
  for (int j = 0; j < 6; ++j) {
    v8f acc = zero8f();
#pragma unroll
    for (int kt = 0; kt < 3; ++kt) {
      v16h a = load_a16(Xs, 96, r0, kt * 32);
      acc = WMMA_F16(a, load_b_pk(Wpk, kt * 6 + j), acc);
    }
    const float bias = b1m[j * 16 + nn];
#pragma unroll
    for (int i = 0; i < 8; ++i)
      Hb[(r0 + mo + i) * 96 + j * 16 + nn] = (_Float16)fmaxf(acc[i] + bias, 0.0f);
  }
  __syncthreads();

  pack_weight(Wpk, W2, 96, 96, tid, 128);
  __syncthreads();

  // ---- out = relu(H @ W2 + b2m)  -> f32 output
#pragma unroll
  for (int j = 0; j < 6; ++j) {
    v8f acc = zero8f();
#pragma unroll
    for (int kt = 0; kt < 3; ++kt) {
      v16h a = load_a16(Hb, 96, r0, kt * 32);
      acc = WMMA_F16(a, load_b_pk(Wpk, kt * 6 + j), acc);
    }
    const float bias = b2m[j * 16 + nn];
#pragma unroll
    for (int i = 0; i < 8; ++i)
      out[(size_t)(tok0 + r0 + mo + i) * Cn + j * 16 + nn] = fmaxf(acc[i] + bias, 0.0f);
  }
}

// ===========================================================================
extern "C" void kernel_launch(void* const* d_in, const int* in_sizes, int n_in,
                              void* d_out, int out_size, void* d_ws, size_t ws_size,
                              hipStream_t stream) {
  const float* x   = (const float*)d_in[0];
  const float* g1  = (const float*)d_in[1];
  const float* b1  = (const float*)d_in[2];
  const float* g2  = (const float*)d_in[3];
  const float* b2  = (const float*)d_in[4];
  const float* Wq  = (const float*)d_in[5];
  const float* bq  = (const float*)d_in[6];
  const float* Wk  = (const float*)d_in[7];
  const float* bk  = (const float*)d_in[8];
  const float* Wv  = (const float*)d_in[9];
  const float* bv  = (const float*)d_in[10];
  const float* Wo  = (const float*)d_in[11];
  const float* bo  = (const float*)d_in[12];
  const float* W1  = (const float*)d_in[13];
  const float* b1m = (const float*)d_in[14];
  const float* W2  = (const float*)d_in[15];
  const float* b2m = (const float*)d_in[16];

  _Float16* xbuf = (_Float16*)d_ws;  // f16 intermediate [B,T,H,W,C] (75.5 MB)

  const int smem1 = 86016;
  const int smem2 = 36352;
  const int smem3 = 43008;
  (void)hipFuncSetAttribute((const void*)k_spatial_msa,
                            hipFuncAttributeMaxDynamicSharedMemorySize, smem1);
  (void)hipFuncSetAttribute((const void*)k_temporal_msa,
                            hipFuncAttributeMaxDynamicSharedMemorySize, smem2);
  (void)hipFuncSetAttribute((const void*)k_mlp,
                            hipFuncAttributeMaxDynamicSharedMemorySize, smem3);

  const int nWin = Bn * Tn * NWIN;      // 6144 windows
  const int nTmp = Bn * HWn / 8;        // 6144 blocks of 8 sequences
  const int nTok = Bn * Tn * HWn / 64;  // 6144 blocks of 64 tokens

  k_spatial_msa<<<nWin, 128, smem1, stream>>>(x, g1, b1, Wq, bq, Wk, bk,
                                              Wv, bv, Wo, bo, xbuf);
  k_temporal_msa<<<nTmp, 128, smem2, stream>>>(Wq, bq, Wk, bk, Wv, bv,
                                               Wo, bo, xbuf);
  k_mlp<<<nTok, 128, smem3, stream>>>(xbuf, g2, b2, W1, b1m, W2, b2m,
                                      (float*)d_out);
}
